// InfSA_42752104465013
// MI455X (gfx1250) — compile-verified, matrix operand order checked by reading, not run on hardware
//
#include <hip/hip_runtime.h>
#include <hip/hip_bf16.h>
#include <math.h>

// ---------------------------------------------------------------------------
// Problem constants (from reference): x [B,C,H,W], B=8, C=256, H=W=64, N=4096
// ---------------------------------------------------------------------------
#define BDIM 8
#define CDIM 256
#define NDIM 4096
#define NUM_STEPS 2
#define GAMMA 0.5f

typedef __attribute__((ext_vector_type(16))) _Float16 v16h;
typedef __attribute__((ext_vector_type(8)))  float    v8f;
typedef __attribute__((ext_vector_type(4)))  unsigned u32x4;
typedef __attribute__((ext_vector_type(8)))  unsigned u32x8;

// D = A(16x32 f16) x B(32x16 f16) + C(16x16 f32)
static __device__ __forceinline__ v8f wmma_f16(v16h a, v16h b, v8f c) {
    return __builtin_amdgcn_wmma_f32_16x16x32_f16(
        /*neg_a=*/false, a, /*neg_b=*/false, b,
        /*c_mod=*/(short)0, c, /*reuse_a=*/false, /*reuse_b=*/false);
}

// Load a 16x32 f16 fragment from an LDS tile stored row-major [row][k]
// (stride in halves = ldk, must be even). Works for A tiles ([m][k]) and for
// B tiles staged K-transposed ([n][k]). Per CDNA5 ISA 16-bit A layout:
//   lanes 0-15 : M=lane,   VGPR j<4 -> K=2j..2j+1,   VGPR j>=4 -> K=16+2(j-4)..
//   lanes 16-31: M=lane-16, same + K offset 8 (low group) / 24 (high group)
static __device__ __forceinline__ v16h load_frag(const _Float16* lds, int row0,
                                                 int ldk, int k0, int lane) {
    union { v16h v; unsigned u[8]; } f;
    const int half = lane >> 4;
    const _Float16* p = lds + (size_t)(row0 + (lane & 15)) * ldk + k0;
#pragma unroll
    for (int j = 0; j < 4; ++j) {
        const int kA = 2 * j + 8 * half;        // K pairs 0..15
        const int kB = 16 + 2 * j + 8 * half;   // K pairs 16..31
        f.u[j]     = *reinterpret_cast<const unsigned*>(p + kA);
        f.u[j + 4] = *reinterpret_cast<const unsigned*>(p + kB);
    }
    return f.v;
}

// ---------------------------------------------------------------------------
// TDM: async DMA of a 2D f16 tile (global, row-major, row stride
// stride0_elems) into LDS at byte offset lds_byte_off, with LDS row padding
// generated by the TDM pad unit (pad_interval/amount codes per D# spec).
// Issue from ONE wave, then s_wait_tensorcnt + workgroup barrier.
// D# group0/group1 bitfields per CDNA5 ISA.
// ---------------------------------------------------------------------------
static __device__ __forceinline__ void tdm_load_tile_f16(
    unsigned lds_byte_off, const _Float16* gsrc,
    unsigned tile_d0, unsigned tile_d1, unsigned stride0_elems,
    unsigned pad_interval_code, unsigned pad_amount_code)
{
    const unsigned long long ga = (unsigned long long)gsrc;
    u32x4 g0;
    g0[0] = 1u;                                        // count=1, user mode
    g0[1] = lds_byte_off;                              // lds_addr [63:32]
    g0[2] = (unsigned)ga;                              // global_addr lo
    g0[3] = (unsigned)((ga >> 32) & 0x1FFFFFFu)        // global_addr hi (56:32)
            | (2u << 30);                              // type=2 ("image")
    u32x8 g1;
    g1[0] = (1u << 16)                                 // data_size = 2 bytes
            | (1u << 20)                               // pad_enable
            | (pad_interval_code << 22)
            | (pad_amount_code << 25);
    g1[1] = (tile_d0 & 0xFFFFu) << 16;                 // tensor_dim0 lo16
    g1[2] = (tile_d0 >> 16) | ((tile_d1 & 0xFFFFu) << 16); // d0 hi / d1 lo
    g1[3] = (tile_d1 >> 16) | (tile_d0 << 16);         // d1 hi | tile_dim0
    g1[4] = (tile_d1 & 0xFFFFu);                       // tile_dim1 (dim2 = 0)
    g1[5] = stride0_elems;                             // tensor_dim0_stride lo
    g1[6] = 0u;
    g1[7] = 0u;
    asm volatile("tensor_load_to_lds %0, %1" :: "s"(g0), "s"(g1) : "memory");
}

// ---------------------------------------------------------------------------
// Kernel 1: Q/K/V projection + row L2-norm.
// Block tile: 64 n-rows x 256 d-cols (full C so the row norm is in-block).
// ---------------------------------------------------------------------------
__global__ __launch_bounds__(256) void qkv_proj_kernel(
    const float* __restrict__ x,
    const float* __restrict__ Wq, const float* __restrict__ Wk,
    const float* __restrict__ Wv,
    _Float16* __restrict__ qh, _Float16* __restrict__ kh,
    _Float16* __restrict__ hh, float* __restrict__ hf)
{
    __shared__ _Float16 At[64][CDIM + 8];   // xt tile  [n_local][c]
    __shared__ _Float16 Bt[CDIM][32 + 8];   // W rows   [d][k_chunk]  (== B^T)
    __shared__ float    rss[64];

    const int tid  = threadIdx.x;
    const int b    = blockIdx.y;
    const int n0   = blockIdx.x * 64;
    const int wsel = blockIdx.z;
    const float* W = (wsel == 0) ? Wq : (wsel == 1) ? Wk : Wv;

    for (int idx = tid; idx < 64 * CDIM; idx += 256) {
        const int nl = idx & 63;
        const int c  = idx >> 6;
        At[nl][c] = (_Float16)x[((size_t)b * CDIM + c) * NDIM + n0 + nl];
    }

    const int lane = tid & 31;
    const int wave = tid >> 5;
    const int wm   = (wave >> 2) * 32;
    const int wn   = (wave & 3) * 64;

    v8f acc[2][4];
#pragma unroll
    for (int i = 0; i < 2; ++i)
#pragma unroll
        for (int j = 0; j < 4; ++j) acc[i][j] = (v8f){0,0,0,0,0,0,0,0};

    for (int kc = 0; kc < CDIM; kc += 32) {
        __syncthreads();
        for (int idx = tid; idx < CDIM * 32; idx += 256) {
            const int d = idx >> 5;
            const int k = idx & 31;
            Bt[d][k] = (_Float16)W[(size_t)d * CDIM + kc + k];
        }
        __syncthreads();

        v16h af[2], bf[4];
#pragma unroll
        for (int i = 0; i < 2; ++i)
            af[i] = load_frag(&At[0][0], wm + 16 * i, CDIM + 8, kc, lane);
#pragma unroll
        for (int j = 0; j < 4; ++j)
            bf[j] = load_frag(&Bt[0][0], wn + 16 * j, 32 + 8, 0, lane);
#pragma unroll
        for (int i = 0; i < 2; ++i)
#pragma unroll
            for (int j = 0; j < 4; ++j)
                acc[i][j] = wmma_f16(af[i], bf[j], acc[i][j]);
    }

    if (tid < 64) rss[tid] = 0.f;
    __syncthreads();
    const int half = lane >> 4, ncol = lane & 15;
#pragma unroll
    for (int i = 0; i < 2; ++i)
#pragma unroll
        for (int j = 0; j < 4; ++j)
#pragma unroll
            for (int r = 0; r < 8; ++r) {
                const float v  = acc[i][j][r];
                const int  row = wm + 16 * i + r + 8 * half;
                atomicAdd(&rss[row], v * v);
            }
    __syncthreads();

#pragma unroll
    for (int i = 0; i < 2; ++i)
#pragma unroll
        for (int j = 0; j < 4; ++j)
#pragma unroll
            for (int r = 0; r < 8; ++r) {
                const int row = wm + 16 * i + r + 8 * half;
                const int col = wn + 16 * j + ncol;
                const float scale = 1.f / fmaxf(sqrtf(rss[row]), 1e-8f);
                const float v = acc[i][j][r] * scale;
                const size_t o = ((size_t)b * NDIM + n0 + row) * CDIM + col;
                if (wsel == 0)      qh[o] = (_Float16)v;
                else if (wsel == 1) kh[o] = (_Float16)v;
                else { hh[o] = (_Float16)v; hf[o] = v; }
            }
}

// ---------------------------------------------------------------------------
// Kernel 2a: zero kv accumulator
// ---------------------------------------------------------------------------
__global__ __launch_bounds__(256) void kv_zero_kernel(float* __restrict__ kvf) {
    const size_t i = (size_t)blockIdx.x * 256 + threadIdx.x;
    if (i < (size_t)BDIM * CDIM * CDIM) kvf[i] = 0.f;
}

// ---------------------------------------------------------------------------
// Kernel 2b: kv[b,c,d] += sum_n k[b,n,c]*h[b,n,d]   (split-K over n, atomics)
// ---------------------------------------------------------------------------
__global__ __launch_bounds__(256) void kv_gemm_kernel(
    const _Float16* __restrict__ kh, const _Float16* __restrict__ hh,
    float* __restrict__ kvf)
{
    __shared__ _Float16 Ak[64][32 + 8];     // K^T tile [c_local][n_chunk]
    __shared__ _Float16 Bh[CDIM][32 + 8];   // H tile, K-transposed [d][n_chunk]

    const int tid   = threadIdx.x;
    const int b     = blockIdx.y;
    const int c0    = blockIdx.x * 64;
    const int nbase = blockIdx.z * (NDIM / 8);

    const int lane = tid & 31;
    const int wave = tid >> 5;
    const int wm   = (wave >> 2) * 32;
    const int wn   = (wave & 3) * 64;

    v8f acc[2][4];
#pragma unroll
    for (int i = 0; i < 2; ++i)
#pragma unroll
        for (int j = 0; j < 4; ++j) acc[i][j] = (v8f){0,0,0,0,0,0,0,0};

    for (int nc = 0; nc < NDIM / 8; nc += 32) {
        __syncthreads();
        const int nn0 = nbase + nc;
        for (int idx = tid; idx < 64 * 32; idx += 256) {
            const int cl = idx & 63;
            const int j  = idx >> 6;
            Ak[cl][j] = kh[((size_t)b * NDIM + nn0 + j) * CDIM + c0 + cl];
        }
        for (int idx = tid; idx < CDIM * 32; idx += 256) {
            const int d = idx & 255;
            const int j = idx >> 8;
            Bh[d][j] = hh[((size_t)b * NDIM + nn0 + j) * CDIM + d];
        }
        __syncthreads();

        v16h af[2], bf[4];
#pragma unroll
        for (int i = 0; i < 2; ++i)
            af[i] = load_frag(&Ak[0][0], wm + 16 * i, 32 + 8, 0, lane);
#pragma unroll
        for (int j = 0; j < 4; ++j)
            bf[j] = load_frag(&Bh[0][0], wn + 16 * j, 32 + 8, 0, lane);
#pragma unroll
        for (int i = 0; i < 2; ++i)
#pragma unroll
            for (int j = 0; j < 4; ++j)
                acc[i][j] = wmma_f16(af[i], bf[j], acc[i][j]);
    }

    const int half = lane >> 4, ncol = lane & 15;
#pragma unroll
    for (int i = 0; i < 2; ++i)
#pragma unroll
        for (int j = 0; j < 4; ++j)
#pragma unroll
            for (int r = 0; r < 8; ++r) {
                const int crow = c0 + wm + 16 * i + r + 8 * half;
                const int dcol = wn + 16 * j + ncol;
                atomicAdd(&kvf[((size_t)b * CDIM + crow) * CDIM + dcol],
                          acc[i][j][r]);
            }
}

// ---------------------------------------------------------------------------
// Kernel 3: column L2-norm of kv (over c), emit f16 kv.
// ---------------------------------------------------------------------------
__global__ __launch_bounds__(256) void kv_norm_kernel(
    const float* __restrict__ kvf, _Float16* __restrict__ kvh)
{
    const int b = blockIdx.x;
    const int d = threadIdx.x;
    const float*    p = kvf + (size_t)b * CDIM * CDIM + d;
    _Float16*       q = kvh + (size_t)b * CDIM * CDIM + d;
    float ss = 0.f;
#pragma unroll 8
    for (int c = 0; c < CDIM; ++c) { const float v = p[(size_t)c * CDIM]; ss += v * v; }
    const float inv = 1.f / (sqrtf(ss) + 1e-8f);
#pragma unroll 8
    for (int c = 0; c < CDIM; ++c)
        q[(size_t)c * CDIM] = (_Float16)(p[(size_t)c * CDIM] * inv);
}

// ---------------------------------------------------------------------------
// Kernel 4: agg = Q * kv ; h = clip(0.5h + 0.5agg, +-10); optional fused LN.
// A tile (64x256 f16, no transpose) staged by the Tensor Data Mover with
// TDM-generated LDS row padding (row = 128 DW -> interval code 6, pad 4 DW
// -> code 3, giving the 256+8 half row stride the fragment loader expects).
// ---------------------------------------------------------------------------
__global__ __launch_bounds__(256) void agg_update_kernel(
    const _Float16* __restrict__ qh, const _Float16* __restrict__ kvh,
    float* __restrict__ hf, _Float16* __restrict__ hh,
    const float* __restrict__ ln_w, const float* __restrict__ ln_b,
    int do_ln)
{
    __shared__ _Float16 At[64][CDIM + 8];   // q tile [n_local][c]  (TDM-filled)
    __shared__ _Float16 Bt[CDIM][32 + 8];   // kv chunk, K-transposed [d][c]
    __shared__ float rsum[64], rsum2[64];

    const int tid = threadIdx.x;
    const int b   = blockIdx.y;
    const int n0  = blockIdx.x * 64;

    const int lane = tid & 31;
    const int wave = tid >> 5;
    const int wm   = (wave >> 2) * 32;
    const int wn   = (wave & 3) * 64;

    // Async DMA the whole A tile via TDM from wave 0, then publish.
    if (wave == 0) {
        tdm_load_tile_f16((unsigned)(size_t)&At[0][0],
                          qh + ((size_t)b * NDIM + n0) * CDIM,
                          /*tile_d0=*/CDIM, /*tile_d1=*/64,
                          /*stride0=*/CDIM,
                          /*pad_interval(128DW)=*/6u, /*pad_amount(4DW)=*/3u);
        __builtin_amdgcn_s_wait_tensorcnt(0);
    }
    __syncthreads();

    v8f acc[2][4];
#pragma unroll
    for (int i = 0; i < 2; ++i)
#pragma unroll
        for (int j = 0; j < 4; ++j) acc[i][j] = (v8f){0,0,0,0,0,0,0,0};

    for (int kc = 0; kc < CDIM; kc += 32) {
        if (kc) __syncthreads();
        for (int idx = tid; idx < CDIM * 32; idx += 256) {
            const int d = idx & 255;
            const int k = idx >> 8;
            Bt[d][k] = kvh[(size_t)b * CDIM * CDIM + (size_t)(kc + k) * CDIM + d];
        }
        __syncthreads();

        v16h af[2], bf[4];
#pragma unroll
        for (int i = 0; i < 2; ++i)
            af[i] = load_frag(&At[0][0], wm + 16 * i, CDIM + 8, kc, lane);
#pragma unroll
        for (int j = 0; j < 4; ++j)
            bf[j] = load_frag(&Bt[0][0], wn + 16 * j, 32 + 8, 0, lane);
#pragma unroll
        for (int i = 0; i < 2; ++i)
#pragma unroll
            for (int j = 0; j < 4; ++j)
                acc[i][j] = wmma_f16(af[i], bf[j], acc[i][j]);
    }
    __syncthreads();

    const int half = lane >> 4, ncol = lane & 15;

#pragma unroll
    for (int i = 0; i < 2; ++i)
#pragma unroll
        for (int j = 0; j < 4; ++j)
#pragma unroll
            for (int r = 0; r < 8; ++r) {
                const int row = wm + 16 * i + r + 8 * half;
                const int col = wn + 16 * j + ncol;
                const size_t o = ((size_t)b * NDIM + n0 + row) * CDIM + col;
                float v = (1.f - GAMMA) * hf[o] + GAMMA * acc[i][j][r];
                v = fminf(10.f, fmaxf(-10.f, v));
                acc[i][j][r] = v;
            }

    if (!do_ln) {
#pragma unroll
        for (int i = 0; i < 2; ++i)
#pragma unroll
            for (int j = 0; j < 4; ++j)
#pragma unroll
                for (int r = 0; r < 8; ++r) {
                    const int row = wm + 16 * i + r + 8 * half;
                    const int col = wn + 16 * j + ncol;
                    const size_t o = ((size_t)b * NDIM + n0 + row) * CDIM + col;
                    hf[o] = acc[i][j][r];
                    hh[o] = (_Float16)acc[i][j][r];
                }
    } else {
        if (tid < 64) { rsum[tid] = 0.f; rsum2[tid] = 0.f; }
        __syncthreads();
#pragma unroll
        for (int i = 0; i < 2; ++i)
#pragma unroll
            for (int j = 0; j < 4; ++j)
#pragma unroll
                for (int r = 0; r < 8; ++r) {
                    const int row = wm + 16 * i + r + 8 * half;
                    const float v = acc[i][j][r];
                    atomicAdd(&rsum[row], v);
                    atomicAdd(&rsum2[row], v * v);
                }
        __syncthreads();
#pragma unroll
        for (int i = 0; i < 2; ++i)
#pragma unroll
            for (int j = 0; j < 4; ++j)
#pragma unroll
                for (int r = 0; r < 8; ++r) {
                    const int row = wm + 16 * i + r + 8 * half;
                    const int col = wn + 16 * j + ncol;
                    const float mu  = rsum[row] * (1.f / CDIM);
                    const float var = rsum2[row] * (1.f / CDIM) - mu * mu;
                    const float inv = rsqrtf(var + 1e-5f);
                    const float v = (acc[i][j][r] - mu) * inv * ln_w[col] + ln_b[col];
                    const size_t o = ((size_t)b * NDIM + n0 + row) * CDIM + col;
                    hh[o] = (_Float16)v;
                }
    }
}

// ---------------------------------------------------------------------------
// Kernel 5: out[b,d,n] = sum_c Wo[d,c]*h[b,n,c] + bo[d]   (output [B,C,N] f32)
// The 256x32 f16 B chunk (no transpose) is staged per k-step by the TDM
// (row = 16 DW -> interval code 3, pad 4 DW -> code 3 => 32+8 half stride).
// ---------------------------------------------------------------------------
__global__ __launch_bounds__(256) void out_gemm_kernel(
    const _Float16* __restrict__ hh, const float* __restrict__ Wo,
    const float* __restrict__ bo, float* __restrict__ out)
{
    __shared__ _Float16 Aw[64][32 + 8];     // Wo tile [d_local][c_chunk]
    __shared__ _Float16 Bx[CDIM][32 + 8];   // h tile [n_local][c]  (TDM-filled)

    const int tid = threadIdx.x;
    const int b   = blockIdx.z;
    const int d0  = blockIdx.y * 64;
    const int n0  = blockIdx.x * 256;

    const int lane = tid & 31;
    const int wave = tid >> 5;
    const int wm   = (wave >> 2) * 32;
    const int wn   = (wave & 3) * 64;

    v8f acc[2][4];
#pragma unroll
    for (int i = 0; i < 2; ++i)
#pragma unroll
        for (int j = 0; j < 4; ++j) acc[i][j] = (v8f){0,0,0,0,0,0,0,0};

    for (int kc = 0; kc < CDIM; kc += 32) {
        __syncthreads();
        for (int idx = tid; idx < 64 * 32; idx += 256) {
            const int k  = idx & 31;
            const int dl = idx >> 5;
            Aw[dl][k] = (_Float16)Wo[(size_t)(d0 + dl) * CDIM + kc + k];
        }
        if (wave == 0) {
            tdm_load_tile_f16((unsigned)(size_t)&Bx[0][0],
                              hh + ((size_t)b * NDIM + n0) * CDIM + kc,
                              /*tile_d0=*/32, /*tile_d1=*/CDIM,
                              /*stride0=*/CDIM,
                              /*pad_interval(16DW)=*/3u, /*pad_amount(4DW)=*/3u);
            __builtin_amdgcn_s_wait_tensorcnt(0);
        }
        __syncthreads();

        v16h af[2], bf[4];
#pragma unroll
        for (int i = 0; i < 2; ++i)
            af[i] = load_frag(&Aw[0][0], wm + 16 * i, 32 + 8, 0, lane);
#pragma unroll
        for (int j = 0; j < 4; ++j)
            bf[j] = load_frag(&Bx[0][0], wn + 16 * j, 32 + 8, 0, lane);
#pragma unroll
        for (int i = 0; i < 2; ++i)
#pragma unroll
            for (int j = 0; j < 4; ++j)
                acc[i][j] = wmma_f16(af[i], bf[j], acc[i][j]);
    }

    const int half = lane >> 4, ncol = lane & 15;
#pragma unroll
    for (int i = 0; i < 2; ++i)
#pragma unroll
        for (int j = 0; j < 4; ++j)
#pragma unroll
            for (int r = 0; r < 8; ++r) {
                const int drow = d0 + wm + 16 * i + r + 8 * half;
                const int ncg  = n0 + wn + 16 * j + ncol;
                out[((size_t)b * CDIM + drow) * NDIM + ncg] =
                    acc[i][j][r] + bo[drow];
            }
}

// ---------------------------------------------------------------------------
// Host launcher
// ---------------------------------------------------------------------------
extern "C" void kernel_launch(void* const* d_in, const int* in_sizes, int n_in,
                              void* d_out, int out_size, void* d_ws, size_t ws_size,
                              hipStream_t stream) {
    const float* x    = (const float*)d_in[0];
    const float* Wq   = (const float*)d_in[1];
    const float* Wk   = (const float*)d_in[2];
    const float* Wv   = (const float*)d_in[3];
    const float* Wo   = (const float*)d_in[4];
    const float* bo   = (const float*)d_in[5];
    const float* ln_w = (const float*)d_in[6];
    const float* ln_b = (const float*)d_in[7];
    float* out = (float*)d_out;

    const size_t NC  = (size_t)BDIM * NDIM * CDIM;   // 8,388,608
    const size_t CC  = (size_t)BDIM * CDIM * CDIM;   //   524,288
    _Float16* qh  = (_Float16*)d_ws;
    _Float16* kh  = qh + NC;
    _Float16* hh  = kh + NC;
    _Float16* kvh = hh + NC;
    float*    hf  = (float*)(kvh + CC);
    float*    kvf = hf + NC;

    qkv_proj_kernel<<<dim3(NDIM / 64, BDIM, 3), 256, 0, stream>>>(
        x, Wq, Wk, Wv, qh, kh, hh, hf);

    for (int step = 0; step < NUM_STEPS; ++step) {
        kv_zero_kernel<<<(unsigned)((CC + 255) / 256), 256, 0, stream>>>(kvf);
        kv_gemm_kernel<<<dim3(CDIM / 64, BDIM, 8), 256, 0, stream>>>(kh, hh, kvf);
        kv_norm_kernel<<<BDIM, 256, 0, stream>>>(kvf, kvh);
        agg_update_kernel<<<dim3(NDIM / 64, BDIM), 256, 0, stream>>>(
            qh, kvh, hf, hh, ln_w, ln_b, (step == NUM_STEPS - 1) ? 1 : 0);
    }

    out_gemm_kernel<<<dim3(NDIM / 256, CDIM / 64, BDIM), 256, 0, stream>>>(
        hh, Wo, bo, out);
}
